// GATNetReduced2LayersLeakyReLU_31628139168052
// MI455X (gfx1250) — matrix operand order repaired and checked by reading, not compiled, not run on hardware
//
#include <hip/hip_runtime.h>
#include <hip/hip_bf16.h>
#include <math.h>

typedef __attribute__((ext_vector_type(16))) __bf16 bf16x16;
typedef __attribute__((ext_vector_type(8)))  __bf16 bf16x8;
typedef __attribute__((ext_vector_type(8)))  float  f32x8;

#define HC   256   // HEADS*OUT_C
#define FIN  512
#define NEG_ATT 0.2f
#define NEG_ACT 0.01f

__device__ __forceinline__ float leakyf(float x, float s) { return x >= 0.f ? x : s * x; }

// ---------------------------------------------------------------------------
// K0: W [512,256] f32  ->  Wt_hi/Wt_lo [256,512] bf16 (transposed, split)
// ---------------------------------------------------------------------------
__global__ void k_convert_w(const float* __restrict__ W,
                            __bf16* __restrict__ WtHi, __bf16* __restrict__ WtLo) {
    int t = blockIdx.x * blockDim.x + threadIdx.x;
    if (t >= FIN * HC) return;
    int k = t >> 8, n = t & 255;
    float v  = W[t];
    __bf16 hi = (__bf16)v;
    __bf16 lo = (__bf16)(v - (float)hi);
    WtHi[n * FIN + k] = hi;
    WtLo[n * FIN + k] = lo;
}

// ---------------------------------------------------------------------------
// K1: xp[N,256] = x[N,512] @ W  via bf16-split WMMA.
// One wave handles one 16-row M-tile and FOUR 16-col N-tiles:
//   - A tile converted to bf16 hi/lo once per K-step, reused by 4 column tiles
//   - 12 WMMAs per K-step across 4 independent accumulator chains
// ---------------------------------------------------------------------------
__global__ void k_gemm_xp(const float* __restrict__ x,
                          const __bf16* __restrict__ WtHi,
                          const __bf16* __restrict__ WtLo,
                          float* __restrict__ xp, int N) {
    const int lane = threadIdx.x & 31;
    const int task = blockIdx.x * 8 + (threadIdx.x >> 5);   // (N/16)*4 tasks
    const int mt  = task >> 2;        // row tile
    const int ntg = task & 3;         // group of 4 col tiles
    if (mt * 16 >= N) return;

    const int rowA = mt * 16 + (lane & 15);
    const int kShA = (lane >> 4) ? 8 : 0;     // A: hi-half lanes hold K+8..15 / +24..31
    const int kShB = (lane >> 4) ? 8 : 0;     // B: hi-half lanes hold K+8..15 / +24..31

    const float* arow = x + (size_t)rowA * FIN;
    const __bf16* bhrow[4];
    const __bf16* blrow[4];
#pragma unroll
    for (int c = 0; c < 4; ++c) {
        int colB = ntg * 64 + c * 16 + (lane & 15);
        bhrow[c] = WtHi + (size_t)colB * FIN;
        blrow[c] = WtLo + (size_t)colB * FIN;
    }

    f32x8 acc[4] = {f32x8{}, f32x8{}, f32x8{}, f32x8{}};

    for (int k0 = 0; k0 < FIN; k0 += 32) {
        // ---- A operand: 16 f32 -> bf16 hi/lo, ISA 16x32 layout ----
        const float4* ap = reinterpret_cast<const float4*>(arow + k0 + kShA);
        float4 f0 = ap[0], f1 = ap[1];
        const float4* ap2 = reinterpret_cast<const float4*>(arow + k0 + kShA + 16);
        float4 f2 = ap2[0], f3 = ap2[1];
        float fv[16] = { f0.x,f0.y,f0.z,f0.w, f1.x,f1.y,f1.z,f1.w,
                         f2.x,f2.y,f2.z,f2.w, f3.x,f3.y,f3.z,f3.w };
        bf16x16 ahi, alo;
#pragma unroll
        for (int i = 0; i < 16; ++i) {
            __bf16 h = (__bf16)fv[i];
            ahi[i] = h;
            alo[i] = (__bf16)(fv[i] - (float)h);
        }
        // ---- B operands: two runs of 8 bf16 per lane (same k order as A), 4 column tiles ----
        bf16x16 bhi[4], blo[4];
#pragma unroll
        for (int c = 0; c < 4; ++c) {
            const bf16x8* bh = reinterpret_cast<const bf16x8*>(bhrow[c] + k0 + kShB);
            const bf16x8* bl = reinterpret_cast<const bf16x8*>(blrow[c] + k0 + kShB);
            bf16x8 bh0 = bh[0], bh1 = bh[2], bl0 = bl[0], bl1 = bl[2];
#pragma unroll
            for (int i = 0; i < 8; ++i) {
                bhi[c][i] = bh0[i]; bhi[c][i + 8] = bh1[i];
                blo[c][i] = bl0[i]; blo[c][i + 8] = bl1[i];
            }
        }
        // ---- 12 WMMAs: 4 independent accumulator chains interleaved ----
#pragma unroll
        for (int c = 0; c < 4; ++c)
            acc[c] = __builtin_amdgcn_wmma_f32_16x16x32_bf16(false, ahi, false, bhi[c], (short)0, acc[c], false, false);
#pragma unroll
        for (int c = 0; c < 4; ++c)
            acc[c] = __builtin_amdgcn_wmma_f32_16x16x32_bf16(false, ahi, false, blo[c], (short)0, acc[c], false, false);
#pragma unroll
        for (int c = 0; c < 4; ++c)
            acc[c] = __builtin_amdgcn_wmma_f32_16x16x32_bf16(false, alo, false, bhi[c], (short)0, acc[c], false, false);
    }
    // C layout: VGPR r -> (M = r + 8*laneHi, N = lane&15)
    const int row0 = mt * 16 + ((lane >> 4) << 3);
#pragma unroll
    for (int c = 0; c < 4; ++c) {
        const int col = ntg * 64 + c * 16 + (lane & 15);
#pragma unroll
        for (int r = 0; r < 8; ++r)
            xp[(size_t)(row0 + r) * HC + col] = acc[c][r];
    }
}

// ---------------------------------------------------------------------------
// K2: attention logits  a_src[n,h], a_dst[n,h]   (wave per node)
// ---------------------------------------------------------------------------
__global__ void k_att(const float* __restrict__ xp,
                      const float* __restrict__ att_src, const float* __restrict__ att_dst,
                      float* __restrict__ a_src, float* __restrict__ a_dst, int N) {
    int node = (blockIdx.x * blockDim.x + threadIdx.x) >> 5;
    int lane = threadIdx.x & 31;
    if (node >= N) return;
    float ps0 = 0, ps1 = 0, pd0 = 0, pd1 = 0;
#pragma unroll
    for (int j = 0; j < 8; ++j) {
        int ch = lane + 32 * j;
        float v  = xp[(size_t)node * HC + ch];
        float as = att_src[ch], ad = att_dst[ch];
        if (j < 4) { ps0 += v * as; pd0 += v * ad; }
        else       { ps1 += v * as; pd1 += v * ad; }
    }
#pragma unroll
    for (int m = 16; m > 0; m >>= 1) {
        ps0 += __shfl_xor(ps0, m, 32); ps1 += __shfl_xor(ps1, m, 32);
        pd0 += __shfl_xor(pd0, m, 32); pd1 += __shfl_xor(pd1, m, 32);
    }
    if (lane == 0) {
        a_src[2 * node] = ps0; a_src[2 * node + 1] = ps1;
        a_dst[2 * node] = pd0; a_dst[2 * node + 1] = pd1;
    }
}

// ---------------------------------------------------------------------------
// CSR build: zero -> histogram -> scan (1 block) -> scatter
// edge e in [0,E): src=ei[e], dst=ei[E+e]; e in [E,E+N): self loop (e-E)
// ---------------------------------------------------------------------------
__global__ void k_zero(int* __restrict__ cnt, int N) {
    int t = blockIdx.x * blockDim.x + threadIdx.x;
    if (t < N) cnt[t] = 0;
}

__global__ void k_hist(const int* __restrict__ ei, int* __restrict__ cnt, int E, int N) {
    int e = blockIdx.x * blockDim.x + threadIdx.x;
    if (e >= E + N) return;
    int d = (e < E) ? ei[E + e] : (e - E);
    atomicAdd(&cnt[d], 1);
}

__global__ void k_scan(const int* __restrict__ cnt, int* __restrict__ rowptr,
                       int* __restrict__ cursor, int N) {
    // single block, 1024 threads, 16 elements/thread (N == 16384)
    __shared__ int part[1024];
    const int tid = threadIdx.x;
    int vals[16], excl[16];
    int run = 0;
#pragma unroll
    for (int t = 0; t < 16; ++t) {
        int idx = tid * 16 + t;
        int v = (idx < N) ? cnt[idx] : 0;
        excl[t] = run; run += v; vals[t] = run;
    }
    part[tid] = run;
    __syncthreads();
    for (int o = 1; o < 1024; o <<= 1) {
        int v = (tid >= o) ? part[tid - o] : 0;
        __syncthreads();
        part[tid] += v;
        __syncthreads();
    }
    int off = (tid > 0) ? part[tid - 1] : 0;
#pragma unroll
    for (int t = 0; t < 16; ++t) {
        int idx = tid * 16 + t;
        if (idx < N) {
            rowptr[idx + 1] = off + vals[t];
            cursor[idx]     = off + excl[t];
        }
    }
    if (tid == 0) rowptr[0] = 0;
}

__global__ void k_scatter(const int* __restrict__ ei, int* __restrict__ cursor,
                          int* __restrict__ csr_src, int E, int N) {
    int e = blockIdx.x * blockDim.x + threadIdx.x;
    if (e >= E + N) return;
    int d = (e < E) ? ei[E + e] : (e - E);
    int s = (e < E) ? ei[e]     : (e - E);
    int pos = atomicAdd(&cursor[d], 1);
    csr_src[pos] = s;
}

// ---------------------------------------------------------------------------
// K5: fused softmax + message aggregation + bias + leaky  (wave per dst node)
// ---------------------------------------------------------------------------
__global__ void k_aggregate(const float* __restrict__ xp,
                            const float* __restrict__ a_src, const float* __restrict__ a_dst,
                            const int* __restrict__ rowptr, const int* __restrict__ csr_src,
                            const float* __restrict__ bias,
                            float* __restrict__ hbuf, int N) {
    const int d    = blockIdx.x * 8 + (threadIdx.x >> 5);
    const int lane = threadIdx.x & 31;
    if (d >= N) return;
    const int beg = rowptr[d], end = rowptr[d + 1];
    const float ad0 = a_dst[2 * d], ad1 = a_dst[2 * d + 1];

    // pass 1: per-head max
    float m0 = -INFINITY, m1 = -INFINITY;
    for (int i = beg + lane; i < end; i += 32) {
        int s = csr_src[i];
        m0 = fmaxf(m0, leakyf(a_src[2 * s]     + ad0, NEG_ATT));
        m1 = fmaxf(m1, leakyf(a_src[2 * s + 1] + ad1, NEG_ATT));
    }
#pragma unroll
    for (int m = 16; m > 0; m >>= 1) {
        m0 = fmaxf(m0, __shfl_xor(m0, m, 32));
        m1 = fmaxf(m1, __shfl_xor(m1, m, 32));
    }
    // pass 2: exp-sum
    float s0 = 0.f, s1 = 0.f;
    for (int i = beg + lane; i < end; i += 32) {
        int s = csr_src[i];
        s0 += expf(leakyf(a_src[2 * s]     + ad0, NEG_ATT) - m0);
        s1 += expf(leakyf(a_src[2 * s + 1] + ad1, NEG_ATT) - m1);
    }
#pragma unroll
    for (int m = 16; m > 0; m >>= 1) {
        s0 += __shfl_xor(s0, m, 32);
        s1 += __shfl_xor(s1, m, 32);
    }
    const float inv0 = 1.f / (s0 + 1e-16f);
    const float inv1 = 1.f / (s1 + 1e-16f);

    // pass 3: weighted accumulate (lane covers channels lane+32j)
    float acc[8] = {0, 0, 0, 0, 0, 0, 0, 0};
    for (int i = beg; i < end; ++i) {
        int s = csr_src[i];
        float a0 = expf(leakyf(a_src[2 * s]     + ad0, NEG_ATT) - m0) * inv0;
        float a1 = expf(leakyf(a_src[2 * s + 1] + ad1, NEG_ATT) - m1) * inv1;
        const float* xr = xp + (size_t)s * HC;
#pragma unroll
        for (int j = 0; j < 8; ++j)
            acc[j] += ((j < 4) ? a0 : a1) * xr[lane + 32 * j];
    }
#pragma unroll
    for (int j = 0; j < 8; ++j) {
        int ch = lane + 32 * j;
        hbuf[(size_t)d * HC + ch] = leakyf(acc[j] + bias[ch], NEG_ACT);
    }
}

// ---------------------------------------------------------------------------
// K6: MLP 256 -> 64 (leaky) -> 3, plus q = |y|^2   (block of 64 per node)
// ---------------------------------------------------------------------------
__global__ void k_mlp(const float* __restrict__ hbuf,
                      const float* __restrict__ W1, const float* __restrict__ b1,
                      const float* __restrict__ W2, const float* __restrict__ b2,
                      float* __restrict__ y, float* __restrict__ q) {
    __shared__ float sh[HC];
    __shared__ float hid[64];
    __shared__ float ys[3];
    const int n = blockIdx.x, tid = threadIdx.x;
    for (int idx = tid; idx < HC; idx += 64) sh[idx] = hbuf[(size_t)n * HC + idx];
    __syncthreads();
    float acc = b1[tid];
    for (int j = 0; j < HC; ++j) acc += sh[j] * W1[j * 64 + tid];
    hid[tid] = leakyf(acc, NEG_ACT);
    __syncthreads();
    if (tid < 3) {
        float o = b2[tid];
#pragma unroll
        for (int j = 0; j < 64; ++j) o += hid[j] * W2[j * 3 + tid];
        ys[tid] = o;
        y[n * 3 + tid] = o;
    }
    __syncthreads();
    if (tid == 0) q[n] = ys[0] * ys[0] + ys[1] * ys[1] + ys[2] * ys[2];
}

// ---------------------------------------------------------------------------
// K7: cdist via gram trick; tile 16 rows x 1024 cols, float4 streaming stores
// ---------------------------------------------------------------------------
__global__ void k_cdist(const float* __restrict__ y, const float* __restrict__ q,
                        float* __restrict__ out, int N) {
    __shared__ float yis[16 * 3];
    __shared__ float qis[16];
    const int tid = threadIdx.x;
    const int j0 = blockIdx.x * 1024 + tid * 4;
    const int i0 = blockIdx.y * 16;
    float yj[4][3], qj[4];
#pragma unroll
    for (int t = 0; t < 4; ++t) {
        int j = j0 + t;
        yj[t][0] = y[j * 3]; yj[t][1] = y[j * 3 + 1]; yj[t][2] = y[j * 3 + 2];
        qj[t] = q[j];
    }
    if (tid < 16) {
        int i = i0 + tid;
        yis[tid * 3] = y[i * 3]; yis[tid * 3 + 1] = y[i * 3 + 1]; yis[tid * 3 + 2] = y[i * 3 + 2];
        qis[tid] = q[i];
    }
    __syncthreads();
#pragma unroll 4
    for (int r = 0; r < 16; ++r) {
        float yi0 = yis[r * 3], yi1 = yis[r * 3 + 1], yi2 = yis[r * 3 + 2], qi = qis[r];
        float4 res;
        float* rp = &res.x;
#pragma unroll
        for (int t = 0; t < 4; ++t) {
            float dot = yi0 * yj[t][0] + yi1 * yj[t][1] + yi2 * yj[t][2];
            float d2 = qi + qj[t] - 2.f * dot;
            d2 = fmaxf(d2, 0.f);
            rp[t] = (d2 > 0.f) ? sqrtf(d2) : 0.f;
        }
        *reinterpret_cast<float4*>(out + (size_t)(i0 + r) * N + j0) = res;
    }
}

// ---------------------------------------------------------------------------
extern "C" void kernel_launch(void* const* d_in, const int* in_sizes, int n_in,
                              void* d_out, int out_size, void* d_ws, size_t ws_size,
                              hipStream_t stream) {
    const float* x       = (const float*)d_in[0];
    const int*   ei      = (const int*)  d_in[1];
    const float* W       = (const float*)d_in[2];
    const float* att_src = (const float*)d_in[3];
    const float* att_dst = (const float*)d_in[4];
    const float* bias    = (const float*)d_in[5];
    const float* W1      = (const float*)d_in[6];
    const float* b1      = (const float*)d_in[7];
    const float* W2      = (const float*)d_in[8];
    const float* b2      = (const float*)d_in[9];
    float* out = (float*)d_out;

    const int N = in_sizes[0] / FIN;   // 16384
    const int E = in_sizes[1] / 2;     // 524288
    const int EP = E + N;              // with self loops

    // workspace carve-up (256B aligned)
    char* w = (char*)d_ws;
    size_t off = 0;
    auto alloc = [&](size_t bytes) -> void* {
        off = (off + 255) & ~(size_t)255;
        void* p = w + off; off += bytes; return p;
    };
    float*  xp     = (float*) alloc((size_t)N * HC * 4);
    float*  hbuf   = (float*) alloc((size_t)N * HC * 4);
    float*  a_src  = (float*) alloc((size_t)N * 2 * 4);
    float*  a_dst  = (float*) alloc((size_t)N * 2 * 4);
    __bf16* WtHi   = (__bf16*)alloc((size_t)HC * FIN * 2);
    __bf16* WtLo   = (__bf16*)alloc((size_t)HC * FIN * 2);
    int*    rowptr = (int*)   alloc((size_t)(N + 1) * 4);
    int*    cursor = (int*)   alloc((size_t)N * 4);
    int*    cnt    = (int*)   alloc((size_t)N * 4);
    int*    csr    = (int*)   alloc((size_t)EP * 4);
    float*  yv     = (float*) alloc((size_t)N * 3 * 4);
    float*  qv     = (float*) alloc((size_t)N * 4);
    (void)ws_size;

    k_convert_w<<<(FIN * HC + 255) / 256, 256, 0, stream>>>(W, WtHi, WtLo);
    // (N/16) M-tiles * 4 N-tile-groups, 8 waves per block
    k_gemm_xp<<<(N / 16) * 4 / 8, 256, 0, stream>>>(x, WtHi, WtLo, xp, N);
    k_att<<<(N * 32 + 255) / 256, 256, 0, stream>>>(xp, att_src, att_dst, a_src, a_dst, N);
    k_zero<<<(N + 255) / 256, 256, 0, stream>>>(cnt, N);
    k_hist<<<(EP + 255) / 256, 256, 0, stream>>>(ei, cnt, E, N);
    k_scan<<<1, 1024, 0, stream>>>(cnt, rowptr, cursor, N);
    k_scatter<<<(EP + 255) / 256, 256, 0, stream>>>(ei, cursor, csr, E, N);
    k_aggregate<<<(N + 7) / 8, 256, 0, stream>>>(xp, a_src, a_dst, rowptr, csr, bias, hbuf, N);
    k_mlp<<<N, 64, 0, stream>>>(hbuf, W1, b1, W2, b2, yv, qv);
    k_cdist<<<dim3(N / 1024, N / 16), 256, 0, stream>>>(yv, qv, out, N);
}
